// NeuralMemory_69458211111434
// MI455X (gfx1250) — matrix operand order repaired and checked by reading, not compiled, not run on hardware
//
#include <hip/hip_runtime.h>
#include <hip/hip_bf16.h>

// ---------------------------------------------------------------------------
// Problem constants (from reference): B=4, L=2048, D=1024, KD=128, NE=32768, TOPK=32
// ---------------------------------------------------------------------------
static constexpr int kD    = 1024;
static constexpr int kKD   = 128;
static constexpr int kNE   = 32768;
static constexpr int kROWS = 4 * 2048;   // B*L = 8192

typedef __bf16 bf16_t;
typedef __attribute__((ext_vector_type(16))) __bf16 v16bf;
typedef __attribute__((ext_vector_type(8)))  __bf16 v8bf;
typedef __attribute__((ext_vector_type(8)))  float  v8f;
typedef __attribute__((ext_vector_type(4)))  int    int4v;
typedef __attribute__((ext_vector_type(8)))  int    int8v;

// ---------------------------------------------------------------------------
// WMMA helper: D = A(16x32 bf16) * B(32x16 bf16) + C(16x16 f32)
// ---------------------------------------------------------------------------
static __device__ __forceinline__ v8f wmma_bf16(v16bf a, v16bf b, v8f c) {
  return __builtin_amdgcn_wmma_f32_16x16x32_bf16(
      /*neg_a=*/false, a, /*neg_b=*/false, b,
      /*c_mod=*/(short)0, c, /*reuse_a=*/false, /*reuse_b=*/false);
}

// A-fragment (16x32, 16-bit): lane L holds row M=L&15.
static __device__ __forceinline__ v16bf load_a_frag(const bf16_t* __restrict__ src,
                                                    int row0, int ld, int k0, int lane) {
  const int m = lane & 15, h = lane >> 4;
  const bf16_t* p = src + (size_t)(row0 + m) * ld + k0 + 16 * h;
  v8bf lo = *(const v8bf*)(p);
  v8bf hi = *(const v8bf*)(p + 8);
  v16bf r;
#pragma unroll
  for (int j = 0; j < 8; ++j) { r[j] = lo[j]; r[j + 8] = hi[j]; }
  return r;
}

// B-fragment (32x16, 16-bit) from "n-major" storage S[n*ldk + k] = B[k][n]:
// lane L holds column N=L&15; elems j -> K = k0 + 16*(L>>4) + j (contiguous 32B)
static __device__ __forceinline__ v16bf load_b_frag_nmajor(const bf16_t* __restrict__ S,
                                                           int n0, int ldk, int k0, int lane) {
  const int n = lane & 15, h = lane >> 4;
  return *(const v16bf*)(S + (size_t)(n0 + n) * ldk + k0 + 16 * h);
}

// Async copy of one wave-private 4KB key tile (16 experts x 128 k, bf16) into LDS.
// 8 x global_load_async_to_lds_b128: 32 lanes x 16B x 8 = 4096B. Tracked by ASYNCcnt.
static __device__ __forceinline__ void stage_keys_async(const bf16_t* __restrict__ gsrc,
                                                        bf16_t* lds_dst, int lane) {
  unsigned l0 = (unsigned)(unsigned long long)lds_dst + (unsigned)lane * 16u;
  unsigned long long g0 = (unsigned long long)gsrc + (unsigned long long)lane * 16ull;
#pragma unroll
  for (int i = 0; i < 8; ++i) {
    asm volatile("global_load_async_to_lds_b128 %0, %1, off"
                 :: "v"(l0 + (unsigned)(i * 512)),
                    "v"(g0 + (unsigned long long)(i * 512))
                 : "memory");
  }
}

// ---------------------------------------------------------------------------
// K0a: fp32 -> bf16 copy
// ---------------------------------------------------------------------------
__global__ void cvt_bf16_kernel(const float* __restrict__ src, bf16_t* __restrict__ dst, long n) {
  long i = (long)blockIdx.x * blockDim.x + threadIdx.x;
  if (i < n) dst[i] = (bf16_t)src[i];
}

// K0b: fp32 (R x C) -> bf16 transposed (C x R)
__global__ void transpose_cvt_kernel(const float* __restrict__ src, bf16_t* __restrict__ dst,
                                     int R, int C) {
  long i = (long)blockIdx.x * blockDim.x + threadIdx.x;
  if (i < (long)R * C) {
    int r = (int)(i / C), c = (int)(i % C);
    dst[(size_t)c * R + r] = (bf16_t)src[i];
  }
}

// ---------------------------------------------------------------------------
// K1: q = hidden @ Wq + bq   (8192x1024)@(1024x128) -> q bf16 (8192x128)
// ---------------------------------------------------------------------------
__global__ __launch_bounds__(256) void qproj_kernel(const bf16_t* __restrict__ hb,
                                                    const bf16_t* __restrict__ wqT,
                                                    const float* __restrict__ bq,
                                                    bf16_t* __restrict__ qb) {
  const int r0 = blockIdx.x * 16;
  const int w = threadIdx.x >> 5, lane = threadIdx.x & 31;
  const int n0 = w * 16;
  v8f c = {0.f, 0.f, 0.f, 0.f, 0.f, 0.f, 0.f, 0.f};
#pragma unroll 4
  for (int s = 0; s < kD / 32; ++s) {
    v16bf a = load_a_frag(hb, r0, kD, 32 * s, lane);
    v16bf b = load_b_frag_nmajor(wqT, n0, kD, 32 * s, lane);
    c = wmma_bf16(a, b, c);
  }
  const int n = lane & 15, h = lane >> 4;
  const float bias = bq[n0 + n];
#pragma unroll
  for (int r = 0; r < 8; ++r)
    qb[(size_t)(r0 + r + 8 * h) * kKD + n0 + n] = (bf16_t)(c[r] + bias);
}

// ---------------------------------------------------------------------------
// K2: fused  sim = q@keys^T  ->  top-32/row  ->  softmax  ->  mem = w@values[idx]
// grid = ROWS/16 blocks, 256 threads (8 waves).
//  - per chunk, each wave async-stages its 16-expert key tile into LDS
//    (double buffered, per-wave private -> only s_wait_asynccnt, no barrier)
//  - 4 chained WMMAs per tile, scores spilled to LDS
//  - top-32 per row lives in a wave's 32 lanes' registers
// ---------------------------------------------------------------------------
__global__ __launch_bounds__(256) void topk_mem_kernel(const bf16_t* __restrict__ qb,
                                                       const bf16_t* __restrict__ keysb,
                                                       const float* __restrict__ values,
                                                       bf16_t* __restrict__ memb) {
  __shared__ float sc[16][132];                                   // scores: 16 x 128 (+pad)
  __shared__ __attribute__((aligned(32))) bf16_t kstage[2][8][16 * kKD];  // 64KB key staging

  const int r0 = blockIdx.x * 16;
  const int w = threadIdx.x >> 5, lane = threadIdx.x & 31;

  // Preload this block's A fragments (16 q-rows, K=128 -> 4 WMMA steps)
  v16bf afr[4];
#pragma unroll
  for (int s = 0; s < 4; ++s) afr[s] = load_a_frag(qb, r0, kKD, 32 * s, lane);

  // Per-lane top-32 state: lane l holds list entry l for rows 2w (idx 0) and 2w+1 (idx 1)
  float lv[2] = {-INFINITY, -INFINITY};
  int   li[2] = {0, 0};
  float thr[2] = {-INFINITY, -INFINITY};

  const int NCH = kNE / 128;  // 256 chunks of 128 experts

  // prime the async pipeline: stage chunk 0
  stage_keys_async(keysb + (size_t)(16 * w) * kKD, &kstage[0][w][0], lane);

  for (int ch = 0; ch < NCH; ++ch) {
    const int e0 = ch * 128;
    const int buf = ch & 1;

    if (ch + 1 < NCH) {
      // stage next chunk, then wait for current one (in-order completion)
      stage_keys_async(keysb + (size_t)(e0 + 128 + 16 * w) * kKD, &kstage[buf ^ 1][w][0], lane);
      asm volatile("s_wait_asynccnt 0x8" ::: "memory");
    } else {
      asm volatile("s_wait_asynccnt 0x0" ::: "memory");
    }

    // --- score tile via WMMA, B fragments read from LDS staging buffer ---
    v8f c = {0.f, 0.f, 0.f, 0.f, 0.f, 0.f, 0.f, 0.f};
    const bf16_t* sb = &kstage[buf][w][0];
#pragma unroll
    for (int s = 0; s < 4; ++s) {
      v16bf b = *(const v16bf*)(sb + (size_t)(lane & 15) * kKD + 32 * s + 16 * (lane >> 4));
      c = wmma_bf16(afr[s], b, c);
    }
    {
      const int n = lane & 15, h = lane >> 4;
#pragma unroll
      for (int r = 0; r < 8; ++r) sc[r + 8 * h][16 * w + n] = c[r];
    }
    __syncthreads();

    // --- top-k maintenance: wave w scans rows 2w and 2w+1 over all 128 cols ---
#pragma unroll
    for (int rr = 0; rr < 2; ++rr) {
      const int lrow = 2 * w + rr;
#pragma unroll
      for (int it = 0; it < 4; ++it) {
        const int col = lane + 32 * it;
        const float v = sc[lrow][col];
        const int gi = e0 + col;
        unsigned mask = (unsigned)__ballot(v > thr[rr]);
        while (mask) {                       // rare after warm-up
          const int src = __ffs(mask) - 1;
          mask &= mask - 1;
          const float cv = __shfl(v, src);
          const int   ci = __shfl(gi, src);
          if (cv <= thr[rr]) continue;
          float mv = lv[rr]; int ml = lane;
#pragma unroll
          for (int off = 16; off > 0; off >>= 1) {
            float ov = __shfl_xor(mv, off);
            int   ol = __shfl_xor(ml, off);
            if (ov < mv || (ov == mv && ol < ml)) { mv = ov; ml = ol; }
          }
          if (cv > mv) {
            if (lane == ml) { lv[rr] = cv; li[rr] = ci; }
            float t = lv[rr];
#pragma unroll
            for (int off = 16; off > 0; off >>= 1) t = fminf(t, __shfl_xor(t, off));
            thr[rr] = t;
          }
        }
      }
    }
    __syncthreads();
  }

  // --- softmax(32) + weighted gather of values rows ---
#pragma unroll
  for (int rr = 0; rr < 2; ++rr) {
    float v = lv[rr];
    float mx = v;
#pragma unroll
    for (int off = 16; off > 0; off >>= 1) mx = fmaxf(mx, __shfl_xor(mx, off));
    float ex = __expf(v - mx);
    float sum = ex;
#pragma unroll
    for (int off = 16; off > 0; off >>= 1) sum += __shfl_xor(sum, off);
    const float wgt = ex / sum;

    float acc[32];
#pragma unroll
    for (int t = 0; t < 32; ++t) acc[t] = 0.f;

    for (int k = 0; k < 32; ++k) {
      const float wk = __shfl(wgt, k);
      const int   ek = __shfl(li[rr], k);
      const float4* vr4 = (const float4*)(values + (size_t)ek * kD);
#pragma unroll
      for (int t4 = 0; t4 < 8; ++t4) {
        float4 f = vr4[lane + 32 * t4];
        acc[4 * t4 + 0] += wk * f.x;
        acc[4 * t4 + 1] += wk * f.y;
        acc[4 * t4 + 2] += wk * f.z;
        acc[4 * t4 + 3] += wk * f.w;
      }
    }
    bf16_t* mo = memb + (size_t)(r0 + 2 * w + rr) * kD;
#pragma unroll
    for (int t4 = 0; t4 < 8; ++t4)
#pragma unroll
      for (int j = 0; j < 4; ++j)
        mo[4 * lane + 128 * t4 + j] = (bf16_t)acc[4 * t4 + j];
  }
}

// ---------------------------------------------------------------------------
// K3: y = mem @ Wo + bo ; gate = sigmoid(hidden.Wg1 + y.Wg2 + bg) ; out = hidden + gate*y
// Per K-step, wave 0 TDM-loads the 1024x32 bf16 WoT tile (64KB) into a double
// buffer; all 8 waves consume it from LDS. TENSORcnt + barriers sequence it.
// ---------------------------------------------------------------------------
__global__ __launch_bounds__(256) void outproj_kernel(const bf16_t* __restrict__ memb,
                                                      const bf16_t* __restrict__ woT,
                                                      const float* __restrict__ bo,
                                                      const float* __restrict__ hidden,
                                                      const float* __restrict__ Wg,
                                                      const float* __restrict__ bg,
                                                      float* __restrict__ out) {
  __shared__ float gpart[16];                                        // gate logit accum
  __shared__ __attribute__((aligned(64))) bf16_t kbuf[2][1024 * 32]; // 2 x 64KB WoT tiles

  const int r0 = blockIdx.x * 16;
  const int w = threadIdx.x >> 5, lane = threadIdx.x & 31;

  if (threadIdx.x < 16) gpart[threadIdx.x] = 0.f;
  __syncthreads();

  // hidden . Wg[0:D] contribution (wave w -> rows 2w, 2w+1)
#pragma unroll
  for (int rr = 0; rr < 2; ++rr) {
    const int row = r0 + 2 * w + rr;
    float s = 0.f;
#pragma unroll 8
    for (int t = 0; t < 32; ++t)
      s += hidden[(size_t)row * kD + lane + 32 * t] * Wg[lane + 32 * t];
#pragma unroll
    for (int off = 16; off > 0; off >>= 1) s += __shfl_xor(s, off);
    if (lane == 0) atomicAdd(&gpart[2 * w + rr], s);
  }

  // TDM issue: D# for a 2D tile, tile_dim0=32 (contiguous k), tile_dim1=1024 rows,
  // tensor_dim0=tensor_dim1=1024, tensor_dim0_stride=1024, data_size=2B.
  auto issue_tdm = [&](int buf, int s) {
    unsigned long long ga = (unsigned long long)(const void*)woT + (unsigned long long)(32 * s) * 2ull;
    unsigned lds_off = (unsigned)(unsigned long long)&kbuf[buf][0];
    int4v g0;
    g0[0] = 1;                                                  // count=1, user mode
    g0[1] = (int)lds_off;                                       // lds_addr
    g0[2] = (int)(ga & 0xffffffffull);                          // global_addr[31:0]
    g0[3] = (int)(((ga >> 32) & 0x01ffffffull) | (2u << 30));   // addr[56:32] | type=2
    int8v g1;
    g1[0] = 1 << 16;        // workgroup_mask=0, data_size=1 (2 bytes)
    g1[1] = 0x0400 << 16;   // tensor_dim0 = 1024 (low 16 @ bits 63:48)
    g1[2] = 0x0400 << 16;   // tensor_dim0 hi=0 | tensor_dim1 = 1024 (low 16 @ bits 95:80)
    g1[3] = 32 << 16;       // tensor_dim1 hi=0 | tile_dim0 = 32
    g1[4] = 1024;           // tile_dim1 = 1024, tile_dim2 = 0
    g1[5] = 1024;           // tensor_dim0_stride low 32 = 1024
    g1[6] = 0;              // stride0 hi | tensor_dim1_stride low (unused, 2D tile)
    g1[7] = 0;
    asm volatile("tensor_load_to_lds %0, %1" :: "s"(g0), "s"(g1) : "memory");
  };

  // GEMM: 16 rows x 128 cols per wave, K=1024, B tiles via TDM double buffer
  v8f cacc[8];
#pragma unroll
  for (int i = 0; i < 8; ++i) cacc[i] = v8f{0.f, 0.f, 0.f, 0.f, 0.f, 0.f, 0.f, 0.f};

  if (w == 0) issue_tdm(0, 0);
  for (int s = 0; s < kD / 32; ++s) {
    const int buf = s & 1;
    if (w == 0) __builtin_amdgcn_s_wait_tensorcnt(0);  // tile `s` resident in kbuf[buf]
    __syncthreads();                                   // publish to all waves
    if (w == 0 && s + 1 < kD / 32) issue_tdm(buf ^ 1, s + 1);  // overlap next DMA

    v16bf a = load_a_frag(memb, r0, kD, 32 * s, lane);
    const bf16_t* sb = &kbuf[buf][0];
#pragma unroll
    for (int i = 0; i < 8; ++i) {
      v16bf b = *(const v16bf*)(sb + (size_t)(128 * w + 16 * i + (lane & 15)) * 32 +
                                16 * (lane >> 4));
      cacc[i] = wmma_bf16(a, b, cacc[i]);
    }
    __syncthreads();  // all waves done with kbuf[buf] before it is overwritten
  }

  const int n = lane & 15, h = lane >> 4;
  float yp[8];
#pragma unroll
  for (int r = 0; r < 8; ++r) yp[r] = 0.f;
#pragma unroll
  for (int i = 0; i < 8; ++i) {
    const int nc = 128 * w + 16 * i + n;
    const float b_o = bo[nc];
    const float wg2 = Wg[kD + nc];
#pragma unroll
    for (int r = 0; r < 8; ++r) {
      float y = cacc[i][r] + b_o;
      cacc[i][r] = y;
      yp[r] += y * wg2;
    }
  }
#pragma unroll
  for (int off = 1; off < 16; off <<= 1)
#pragma unroll
    for (int r = 0; r < 8; ++r) yp[r] += __shfl_xor(yp[r], off);
  if (n == 0)
#pragma unroll
    for (int r = 0; r < 8; ++r) atomicAdd(&gpart[r + 8 * h], yp[r]);
  __syncthreads();

  const float bgv = bg[0];
#pragma unroll
  for (int i = 0; i < 8; ++i) {
    const int nc = 128 * w + 16 * i + n;
#pragma unroll
    for (int r = 0; r < 8; ++r) {
      const int m = r + 8 * h;
      const float g = 1.f / (1.f + __expf(-(gpart[m] + bgv)));
      const size_t idx = (size_t)(r0 + m) * kD + nc;
      out[idx] = hidden[idx] + g * cacc[i][r];
    }
  }
}

// ---------------------------------------------------------------------------
// Host-side launcher
// ---------------------------------------------------------------------------
extern "C" void kernel_launch(void* const* d_in, const int* in_sizes, int n_in,
                              void* d_out, int out_size, void* d_ws, size_t ws_size,
                              hipStream_t stream) {
  const float* hidden = (const float*)d_in[0];  // (B,L,D)
  const float* Wq     = (const float*)d_in[1];  // (D,KD)
  const float* bq     = (const float*)d_in[2];  // (KD)
  const float* keys   = (const float*)d_in[3];  // (NE,KD)
  const float* values = (const float*)d_in[4];  // (NE,D)
  const float* Wo     = (const float*)d_in[5];  // (D,D)
  const float* bo     = (const float*)d_in[6];  // (D)
  const float* Wg     = (const float*)d_in[7];  // (2D,1)
  const float* bg     = (const float*)d_in[8];  // (1)
  float* out = (float*)d_out;

  char* ws = (char*)d_ws;
  size_t off = 0;
  auto take = [&](size_t bytes) -> char* {
    char* p = ws + off;
    off = (off + bytes + 255) & ~(size_t)255;
    return p;
  };
  bf16_t* hb   = (bf16_t*)take(sizeof(bf16_t) * (size_t)kROWS * kD);   // hidden bf16
  bf16_t* wqT  = (bf16_t*)take(sizeof(bf16_t) * (size_t)kKD * kD);     // Wq^T (n-major)
  bf16_t* kb   = (bf16_t*)take(sizeof(bf16_t) * (size_t)kNE * kKD);    // keys bf16
  bf16_t* woT  = (bf16_t*)take(sizeof(bf16_t) * (size_t)kD * kD);      // Wo^T (n-major)
  bf16_t* qb   = (bf16_t*)take(sizeof(bf16_t) * (size_t)kROWS * kKD);  // q bf16
  bf16_t* memb = (bf16_t*)take(sizeof(bf16_t) * (size_t)kROWS * kD);   // mem bf16
  (void)ws_size; (void)in_sizes; (void)n_in; (void)out_size;

  const int T = 256;
  cvt_bf16_kernel<<<(int)(((size_t)kROWS * kD + T - 1) / T), T, 0, stream>>>(hidden, hb, (long)kROWS * kD);
  transpose_cvt_kernel<<<(int)(((size_t)kD * kKD + T - 1) / T), T, 0, stream>>>(Wq, wqT, kD, kKD);
  cvt_bf16_kernel<<<(int)(((size_t)kNE * kKD + T - 1) / T), T, 0, stream>>>(keys, kb, (long)kNE * kKD);
  transpose_cvt_kernel<<<(int)(((size_t)kD * kD + T - 1) / T), T, 0, stream>>>(Wo, woT, kD, kD);

  qproj_kernel<<<kROWS / 16, T, 0, stream>>>(hb, wqT, bq, qb);
  topk_mem_kernel<<<kROWS / 16, T, 0, stream>>>(qb, kb, values, memb);
  outproj_kernel<<<kROWS / 16, T, 0, stream>>>(memb, woT, bo, hidden, Wg, bg, out);
}
